// EnhancementGenerator_29695403884593
// MI455X (gfx1250) — matrix operand order repaired and checked by reading, not compiled, not run on hardware
//
#include <hip/hip_runtime.h>
#include <hip/hip_bf16.h>
#include <cstdint>

// ---------------------------------------------------------------------------
// Model constants
// ---------------------------------------------------------------------------
#define HSZ   40        // hidden size
#define G3    120       // 3*H
#define GPAD  128       // padded gate width
#define FIN   257       // input features / output bins
#define TT    1000      // timesteps
#define BB    64        // batch
#define MTOT  (TT*BB)   // 64000 rows
#define KIN0  257
#define KP0   288       // 257 padded to mult of 32
#define KIN1  40
#define KP1   64
#define KANK  720       // 80 silu + 80*8 spline features
#define KANKP 736       // padded to mult of 32 (23 chunks)
#define NGL   80
#define NGLP  80
#define NLIN  257
#define NLINP 272

typedef _Float16 f16;
typedef __attribute__((ext_vector_type(16))) _Float16 v16h;
typedef __attribute__((ext_vector_type(8)))  _Float16 v8h;
typedef __attribute__((ext_vector_type(8)))  float    v8f;

__device__ __forceinline__ v8f wmma_f16(v16h a, v16h b, v8f c) {
  return __builtin_amdgcn_wmma_f32_16x16x32_f16(false, a, false, b, (short)0, c,
                                                false, false);
}

__device__ __forceinline__ float sigmoidf_(float x) {
  return 1.0f / (1.0f + __expf(-x));
}

// ---------------------------------------------------------------------------
// Pack wih (N x K, row-major) -> f16 Wp (Kpad x Npad), Wp[k][n] = wih[n][k]
// Zero-pads k >= K and n >= N (the GEMM relies on the k-padding being zero).
// ---------------------------------------------------------------------------
__global__ void pack_w_kernel(const float* __restrict__ wih, f16* __restrict__ Wp,
                              int K, int N, int Kpad, int Npad) {
  int idx = blockIdx.x * blockDim.x + threadIdx.x;
  int total = Kpad * Npad;
  if (idx >= total) return;
  int k = idx / Npad, n = idx % Npad;
  float v = (k < K && n < N) ? wih[(size_t)n * K + k] : 0.0f;
  Wp[idx] = (f16)v;
}

// ---------------------------------------------------------------------------
// Pack KAN weights: combined feature matrix (KANKP x Npad) f16.
// row r < 80           : base_w[n][r]
// 80 <= r < 720        : spline_w[n][i][k] * scaler[n][i],  r = 80 + i*8 + k
// r >= 720 or n >= N   : 0
// ---------------------------------------------------------------------------
__global__ void pack_kan_w_kernel(const float* __restrict__ base_w,
                                  const float* __restrict__ spline_w,
                                  const float* __restrict__ scaler,
                                  f16* __restrict__ Wp, int N, int Npad) {
  int idx = blockIdx.x * blockDim.x + threadIdx.x;
  int total = KANKP * Npad;
  if (idx >= total) return;
  int r = idx / Npad, n = idx % Npad;
  float v = 0.0f;
  if (n < N) {
    if (r < 80) {
      v = base_w[(size_t)n * 80 + r];
    } else if (r < KANK) {
      int q = r - 80;
      int i = q >> 3, kk = q & 7;
      v = spline_w[(((size_t)n * 80) + i) * 8 + kk] * scaler[(size_t)n * 80 + i];
    }
  }
  Wp[idx] = (f16)v;
}

// ---------------------------------------------------------------------------
// Input-gate GEMM: Gi[m][n] = sum_k X(m,k) * Wp[k][n] + bih[n]
//   m = t*64 + b ; xmode 0: X row-major (m, K)
//                  xmode 1: X = x[b][t][k]       (forward)
//                  xmode 2: X = x[b][T-1-t][k]   (time reversed)
// One block: 64 rows x 128 cols, 8 waves, each wave owns one 16-col tile.
// A staging: each thread owns 8 contiguous K of one row (vector ds_store_b128).
// K-tail uses index clamping (values beyond K hit zero weight rows), so the
// tail path is branch-free data flow instead of per-element exec masking.
// ---------------------------------------------------------------------------
__global__ __launch_bounds__(256) void gemm_gi_kernel(
    const float* __restrict__ X, const f16* __restrict__ Wp,
    const float* __restrict__ bih, float* __restrict__ Gi,
    int K, int Kpad, int xmode, int T) {
  __shared__ f16 As[64][32];
  __shared__ f16 Bs[32][GPAD];

  const int tid  = threadIdx.x;
  const int lane = tid & 31;
  const int wave = tid >> 5;     // 0..7 -> column tile
  const int nt   = wave;
  const int mBase = blockIdx.x * 64;

  // A staging assignment: row ar, k-offset ac (8 contiguous halves)
  const int ar = tid >> 2;          // 0..63
  const int ac = (tid & 3) * 8;     // 0,8,16,24
  const float* arow;
  {
    int m = mBase + ar;
    if (xmode == 0) {
      arow = X + (size_t)m * K;
    } else {
      int t = m >> 6, b = m & 63;
      int tt = (xmode == 2) ? (T - 1 - t) : t;
      arow = X + ((size_t)b * T + tt) * K;
    }
  }

  v8f acc0 = {}, acc1 = {}, acc2 = {}, acc3 = {};

  const int nchunks = Kpad / 32;
  for (int kk = 0; kk < nchunks; ++kk) {
    __syncthreads();
    const int k0 = kk * 32;
    // Stage A chunk (64x32) as f16: one v8h group per thread
    {
      const int kg = k0 + ac;
      v8h h8;
      if (kg + 8 <= K) {
        #pragma unroll
        for (int i = 0; i < 8; ++i) h8[i] = (f16)arow[kg + i];
      } else {
        // clamped indices: in-range k get correct values; k >= K get a legal
        // (arbitrary) value that multiplies a zero weight row
        #pragma unroll
        for (int i = 0; i < 8; ++i) {
          int kc = kg + i; kc = (kc < K - 1) ? kc : (K - 1);
          h8[i] = (f16)arow[kc];
        }
      }
      *(v8h*)&As[ar][ac] = h8;
    }
    // Stage B chunk (32x128): 16 contiguous halves (32B) per thread
    {
      const f16* src = Wp + (size_t)k0 * GPAD + tid * 16;
      v8h b0, b1;
      #pragma unroll
      for (int i = 0; i < 8; ++i) { b0[i] = src[i]; b1[i] = src[8 + i]; }
      f16* dst = &Bs[0][0] + tid * 16;
      *(v8h*)dst = b0;
      *(v8h*)(dst + 8) = b1;
    }
    // prefetch next weight chunk into caches
    if (k0 + 32 < Kpad)
      __builtin_prefetch(Wp + (size_t)(k0 + 32) * GPAD + tid * 16, 0, 1);
    __syncthreads();

    // B fragment (32x16): lane%16 <-> N, halves <-> K in table order (8*(lane/16) + h, +8 for h>=8)
    v16h bf;
    #pragma unroll
    for (int h = 0; h < 16; ++h) bf[h] = Bs[(lane >> 4) * 8 + ((h < 8) ? h : h + 8)][nt * 16 + (lane & 15)];

    const int mrow = lane & 15;
    const int koff = (lane < 16) ? 0 : 8;
    #pragma unroll
    for (int mt = 0; mt < 4; ++mt) {
      v16h af;
      #pragma unroll
      for (int h = 0; h < 16; ++h) {
        int kc = koff + ((h < 8) ? h : h + 8);
        af[h] = As[mt * 16 + mrow][kc];
      }
      if      (mt == 0) acc0 = wmma_f16(af, bf, acc0);
      else if (mt == 1) acc1 = wmma_f16(af, bf, acc1);
      else if (mt == 2) acc2 = wmma_f16(af, bf, acc2);
      else              acc3 = wmma_f16(af, bf, acc3);
    }
  }

  const int ncol = nt * 16 + (lane & 15);
  const int mr8  = (lane >> 4) * 8;
  const float bv = (ncol < G3) ? bih[ncol] : 0.0f;
  #pragma unroll
  for (int mt = 0; mt < 4; ++mt) {
    v8f c = (mt == 0) ? acc0 : (mt == 1) ? acc1 : (mt == 2) ? acc2 : acc3;
    #pragma unroll
    for (int v = 0; v < 8; ++v) {
      int m = mBase + mt * 16 + mr8 + v;
      float val = (ncol < G3) ? (c[v] + bv) : 0.0f;
      Gi[(size_t)m * GPAD + ncol] = val;
    }
  }
}

// ---------------------------------------------------------------------------
// Persistent GRU recurrence. gridDim.x == 2 (two directions in parallel).
// h, whh, bhh, gate staging all live in LDS; gi streams from L2.
// Writes h sequence (T*64 x 40) and/or concat buffer (T*64 x 80).
// ---------------------------------------------------------------------------
__global__ __launch_bounds__(256) void gru_rec_kernel(
    const float* giA, const float* whhA, const float* bhhA,
    float* hseqA, float* catA, int coffA,
    const float* giB, const float* whhB, const float* bhhB,
    float* hseqB, float* catB, int coffB, int T) {
  const float* Gi;  const float* whh; const float* bhh;
  float* hseq; float* cat; int coff;
  if (blockIdx.x == 0) { Gi = giA; whh = whhA; bhh = bhhA; hseq = hseqA; cat = catA; coff = coffA; }
  else                 { Gi = giB; whh = whhB; bhh = bhhB; hseq = hseqB; cat = catB; coff = coffB; }

  __shared__ float hS[BB][HSZ];
  __shared__ float wS[G3][HSZ];
  __shared__ float bS[G3];
  __shared__ float ghS[BB][G3];

  const int tid = threadIdx.x;
  for (int i = tid; i < BB * HSZ; i += 256) hS[i / HSZ][i % HSZ] = 0.0f;
  for (int i = tid; i < G3 * HSZ; i += 256) wS[i / HSZ][i % HSZ] = whh[i];
  for (int i = tid; i < G3; i += 256) bS[i] = bhh[i];
  __syncthreads();

  for (int t = 0; t < T; ++t) {
    // gh = h @ whh^T + bhh  (64 x 120, dot-40)
    for (int idx = tid; idx < BB * G3; idx += 256) {
      int b = idx / G3, o = idx % G3;
      float s = bS[o];
      #pragma unroll
      for (int j = 0; j < HSZ; ++j) s += hS[b][j] * wS[o][j];
      ghS[b][o] = s;
    }
    __syncthreads();
    const float* gt = Gi + (size_t)t * BB * GPAD;
    for (int idx = tid; idx < BB * HSZ; idx += 256) {
      int b = idx / HSZ, j = idx % HSZ;
      const float* gib = gt + (size_t)b * GPAD;
      float r  = sigmoidf_(gib[j]        + ghS[b][j]);
      float z  = sigmoidf_(gib[HSZ + j]  + ghS[b][HSZ + j]);
      float nn = tanhf(gib[2 * HSZ + j] + r * ghS[b][2 * HSZ + j]);
      float hn = (1.0f - z) * nn + z * hS[b][j];
      hS[b][j] = hn;
      if (hseq) hseq[((size_t)t * BB + b) * HSZ + j] = hn;
      if (cat)  cat[((size_t)t * BB + b) * 80 + coff + j] = hn;
    }
    __syncthreads();
  }
}

// ---------------------------------------------------------------------------
// Fused KAN layer: per row build 720 features (silu + B-spline bases) in LDS
// as f16, then WMMA against prepacked weights. 64 rows x 64 cols per block.
//   mode 0: Out[m*80 + n] = y                      (gl layer)
//   mode 1: Out[b*T*257 + t*257 + n] = 1.2*sigmoid(slope[n]*y)  (final layer)
// ---------------------------------------------------------------------------
__global__ __launch_bounds__(256) void kan_wmma_kernel(
    const float* __restrict__ Xrows, const f16* __restrict__ Wp,
    int N, int Npad, const float* __restrict__ slope,
    float* __restrict__ Out, int mode, int T) {
  __shared__ f16 feat[64][KANKP];
  __shared__ f16 Ws[32][64];

  const int tid  = threadIdx.x;
  const int lane = tid & 31;
  const int wave = tid >> 5;
  const int mBase   = blockIdx.x * 64;
  const int colBase = blockIdx.y * 64;

  // ---- feature construction: silu + Cox-de Boor B-spline bases (order 3) ----
  for (int idx = tid; idx < 64 * 80; idx += 256) {
    int r = idx / 80, i = idx % 80;
    float x = Xrows[((size_t)(mBase + r)) * 80 + i];
    feat[r][i] = (f16)(x * (1.0f / (1.0f + __expf(-x))));
    float b[11];
    #pragma unroll
    for (int j = 0; j < 11; ++j) {
      float gj = 0.4f * (float)j - 2.2f;
      b[j] = (x >= gj && x < gj + 0.4f) ? 1.0f : 0.0f;
    }
    #pragma unroll
    for (int p = 1; p <= 3; ++p) {
      float invd = 1.0f / (0.4f * (float)p);
      #pragma unroll
      for (int j = 0; j < 10; ++j) {
        if (j < 11 - p) {
          float gj   = 0.4f * (float)j - 2.2f;
          float gjp1 = gj + 0.4f * (float)(p + 1);
          b[j] = (x - gj) * invd * b[j] + (gjp1 - x) * invd * b[j + 1];
        }
      }
    }
    v8h h8;
    #pragma unroll
    for (int k = 0; k < 8; ++k) h8[k] = (f16)b[k];
    *(v8h*)&feat[r][80 + i * 8] = h8;   // 16B-aligned (row stride 1472B)
  }
  // zero pad rows 720..735 (v8h groups)
  {
    v8h z = {};
    for (int g = tid; g < 64 * 2; g += 256) {   // 64 rows * 16 halves / 8
      int r = g >> 1, c0 = (g & 1) * 8;
      *(v8h*)&feat[r][KANK + c0] = z;
    }
  }

  // ---- tiling: 4 mt x 4 nt tiles, wave w -> (mt=w>>2, nt=w&3) and mt+2 ----
  const int nt  = wave & 3;
  const int mt0 = wave >> 2;
  const int mt1 = mt0 + 2;
  v8f acc0 = {}, acc1 = {};

  // Ws staging assignment: one v8h group per thread
  const int wr = tid >> 3;          // 0..31
  const int wc = (tid & 7) * 8;     // 0..56

  for (int kk = 0; kk < KANKP / 32; ++kk) {
    __syncthreads();
    const int k0 = kk * 32;
    {
      const int nc0 = colBase + wc;
      v8h h8 = {};
      if (nc0 + 8 <= Npad) {
        const f16* src = Wp + (size_t)(k0 + wr) * Npad + nc0;
        #pragma unroll
        for (int i = 0; i < 8; ++i) h8[i] = src[i];
      }
      *(v8h*)&Ws[wr][wc] = h8;
    }
    if (k0 + 32 < KANKP)
      __builtin_prefetch(Wp + (size_t)(k0 + 32 + wr) * Npad + colBase + wc, 0, 1);
    __syncthreads();

    v16h bf;
    #pragma unroll
    for (int h = 0; h < 16; ++h) bf[h] = Ws[(lane >> 4) * 8 + ((h < 8) ? h : h + 8)][nt * 16 + (lane & 15)];

    const int mrow = lane & 15;
    const int koff = (lane < 16) ? 0 : 8;
    v16h af0, af1;
    #pragma unroll
    for (int h = 0; h < 16; ++h) {
      int kc = k0 + koff + ((h < 8) ? h : h + 8);
      af0[h] = feat[mt0 * 16 + mrow][kc];
      af1[h] = feat[mt1 * 16 + mrow][kc];
    }
    acc0 = wmma_f16(af0, bf, acc0);
    acc1 = wmma_f16(af1, bf, acc1);
  }

  const int ncol = colBase + nt * 16 + (lane & 15);
  const int mr8  = (lane >> 4) * 8;
  #pragma unroll
  for (int pass = 0; pass < 2; ++pass) {
    v8f c = pass ? acc1 : acc0;
    int mt = pass ? mt1 : mt0;
    #pragma unroll
    for (int v = 0; v < 8; ++v) {
      int m = mBase + mt * 16 + mr8 + v;
      float val = c[v];
      if (ncol < N) {
        if (mode == 0) {
          Out[(size_t)m * 80 + ncol] = val;
        } else {
          float y = 1.2f * (1.0f / (1.0f + __expf(-slope[ncol] * val)));
          int t = m >> 6, b = m & 63;
          Out[((size_t)b * T + t) * (size_t)NLIN + ncol] = y;
        }
      }
    }
  }
}

// ---------------------------------------------------------------------------
// Host launch
// ---------------------------------------------------------------------------
extern "C" void kernel_launch(void* const* d_in, const int* in_sizes, int n_in,
                              void* d_out, int out_size, void* d_ws, size_t ws_size,
                              hipStream_t stream) {
  (void)in_sizes; (void)n_in; (void)out_size; (void)ws_size;
  const float* x        = (const float*)d_in[0];
  // d_in[1] = lengths (unused, as in reference)
  const float* wih_f0 = (const float*)d_in[2];
  const float* whh_f0 = (const float*)d_in[3];
  const float* bih_f0 = (const float*)d_in[4];
  const float* bhh_f0 = (const float*)d_in[5];
  const float* wih_f1 = (const float*)d_in[6];
  const float* whh_f1 = (const float*)d_in[7];
  const float* bih_f1 = (const float*)d_in[8];
  const float* bhh_f1 = (const float*)d_in[9];
  const float* wih_b0 = (const float*)d_in[10];
  const float* whh_b0 = (const float*)d_in[11];
  const float* bih_b0 = (const float*)d_in[12];
  const float* bhh_b0 = (const float*)d_in[13];
  const float* wih_b1 = (const float*)d_in[14];
  const float* whh_b1 = (const float*)d_in[15];
  const float* bih_b1 = (const float*)d_in[16];
  const float* bhh_b1 = (const float*)d_in[17];
  const float* gl_base_w   = (const float*)d_in[18];
  const float* gl_spline_w = (const float*)d_in[19];
  const float* gl_scaler   = (const float*)d_in[20];
  const float* lin_base_w   = (const float*)d_in[21];
  const float* lin_spline_w = (const float*)d_in[22];
  const float* lin_scaler   = (const float*)d_in[23];
  const float* slope        = (const float*)d_in[24];
  float* out = (float*)d_out;

  // Workspace carving (256-byte aligned)
  char* ws = (char*)d_ws;
  size_t off = 0;
  auto carve = [&](size_t bytes) -> char* {
    char* p = ws + off;
    off = (off + bytes + 255) & ~(size_t)255;
    return p;
  };
  float* gi_a  = (float*)carve((size_t)MTOT * GPAD * 4);   // 32.8 MB (reused layer0/1)
  float* gi_b  = (float*)carve((size_t)MTOT * GPAD * 4);
  float* h0f   = (float*)carve((size_t)MTOT * HSZ * 4);    // 10.2 MB
  float* h0b   = (float*)carve((size_t)MTOT * HSZ * 4);
  float* catb  = (float*)carve((size_t)MTOT * 80 * 4);     // 20.5 MB
  float* glout = (float*)carve((size_t)MTOT * 80 * 4);
  f16* Wf0p = (f16*)carve((size_t)KP0 * GPAD * 2);
  f16* Wb0p = (f16*)carve((size_t)KP0 * GPAD * 2);
  f16* Wf1p = (f16*)carve((size_t)KP1 * GPAD * 2);
  f16* Wb1p = (f16*)carve((size_t)KP1 * GPAD * 2);
  f16* Wgl  = (f16*)carve((size_t)KANKP * NGLP * 2);
  f16* Wlin = (f16*)carve((size_t)KANKP * NLINP * 2);

  // --- pack weights ---
  {
    int tot0 = KP0 * GPAD, tot1 = KP1 * GPAD;
    pack_w_kernel<<<(tot0 + 255) / 256, 256, 0, stream>>>(wih_f0, Wf0p, KIN0, G3, KP0, GPAD);
    pack_w_kernel<<<(tot0 + 255) / 256, 256, 0, stream>>>(wih_b0, Wb0p, KIN0, G3, KP0, GPAD);
    pack_w_kernel<<<(tot1 + 255) / 256, 256, 0, stream>>>(wih_f1, Wf1p, KIN1, G3, KP1, GPAD);
    pack_w_kernel<<<(tot1 + 255) / 256, 256, 0, stream>>>(wih_b1, Wb1p, KIN1, G3, KP1, GPAD);
    int totg = KANKP * NGLP, totl = KANKP * NLINP;
    pack_kan_w_kernel<<<(totg + 255) / 256, 256, 0, stream>>>(gl_base_w, gl_spline_w, gl_scaler, Wgl, NGL, NGLP);
    pack_kan_w_kernel<<<(totl + 255) / 256, 256, 0, stream>>>(lin_base_w, lin_spline_w, lin_scaler, Wlin, NLIN, NLINP);
  }

  const int mblocks = MTOT / 64;   // 1000

  // --- layer 0 input-gate GEMMs (forward / time-reversed) ---
  gemm_gi_kernel<<<mblocks, 256, 0, stream>>>(x, Wf0p, bih_f0, gi_a, KIN0, KP0, 1, TT);
  gemm_gi_kernel<<<mblocks, 256, 0, stream>>>(x, Wb0p, bih_b0, gi_b, KIN0, KP0, 2, TT);

  // --- layer 0 recurrences (both directions, one block each) ---
  gru_rec_kernel<<<2, 256, 0, stream>>>(
      gi_a, whh_f0, bhh_f0, h0f, (float*)nullptr, 0,
      gi_b, whh_b0, bhh_b0, h0b, (float*)nullptr, 0, TT);

  // --- layer 1 input-gate GEMMs from h0 sequences (buffers reused) ---
  gemm_gi_kernel<<<mblocks, 256, 0, stream>>>(h0f, Wf1p, bih_f1, gi_a, KIN1, KP1, 0, TT);
  gemm_gi_kernel<<<mblocks, 256, 0, stream>>>(h0b, Wb1p, bih_b1, gi_b, KIN1, KP1, 0, TT);

  // --- layer 1 recurrences: write concat(hf1, hb1) directly ---
  gru_rec_kernel<<<2, 256, 0, stream>>>(
      gi_a, whh_f1, bhh_f1, (float*)nullptr, catb, 0,
      gi_b, whh_b1, bhh_b1, (float*)nullptr, catb, HSZ, TT);

  // --- KAN gl layer (80 -> 80) ---
  {
    dim3 g(mblocks, (NGLP + 63) / 64);
    kan_wmma_kernel<<<g, 256, 0, stream>>>(catb, Wgl, NGL, NGLP, (const float*)nullptr,
                                           glout, 0, TT);
  }
  // --- KAN lin layer (80 -> 257) + learnable sigmoid, scattered to (B,T,257) ---
  {
    dim3 g(mblocks, (NLINP + 63) / 64 + 1);  // 6 column groups; 5 cover 272 padded cols, the 6th writes nothing
    kan_wmma_kernel<<<g, 256, 0, stream>>>(glout, Wlin, NLIN, NLINP, slope,
                                           out, 1, TT);
  }
}
